// GAT_Net_86079734546934
// MI455X (gfx1250) — matrix operand order repaired and checked by reading, not compiled, not run on hardware
//
#include <hip/hip_runtime.h>
#include <hip/hip_bf16.h>
#include <stdint.h>

// ---------------------------------------------------------------------------
// GAT layer for MI455X (gfx1250), flash-softmax fused, WMMA everywhere.
//   N=32 batches, V=1024 nodes, F_in=128, H=64, neg_slope=0.01
// Pass 1: h = feat @ W^T via v_wmma_f32_16x16x4_f32 (exact fp32),
//         emits s_src/s_dst (fp32) and h repacked to bf16 in WMMA-B layout.
// Pass 2: out[i,:] = (sum_j p_ij * h_j) / (sum_j p_ij),
//         p_ij = exp(lrelu(s_i + t_j) - m_i), m_i = lrelu(s_i + max_j t_j)
//         via v_wmma_f32_16x16x32_bf16, denominator via WMMA with B=ones.
// ---------------------------------------------------------------------------

typedef __attribute__((ext_vector_type(2)))  float  v2f;
typedef __attribute__((ext_vector_type(8)))  float  v8f;
typedef __attribute__((ext_vector_type(16))) __bf16 v16bf;

#define NEG_SLOPE 0.01f
#define NB   32      // batches
#define NV   1024    // nodes
#define FIN  128     // input features
#define NH   64      // hidden
#define JT   32      // j-tile == K of bf16 WMMA
#define NJT  (NV / JT)

// round-to-nearest-even f32 -> bf16, packed pair into one dword
static __device__ __forceinline__ uint32_t pack_bf16(float lo, float hi) {
    uint32_t a = __float_as_uint(lo);
    uint32_t b = __float_as_uint(hi);
    uint32_t ar = (a + 0x7FFFu + ((a >> 16) & 1u)) >> 16;
    uint32_t br = (b + 0x7FFFu + ((b >> 16) & 1u)) >> 16;
    return ar | (br << 16);
}

static __device__ __forceinline__ float lrelu(float x) {
    return x >= 0.0f ? x : NEG_SLOPE * x;
}

union alignas(32) BfTile { uint32_t u[8]; v16bf v; };

// ---------------------------------------------------------------------------
// Pass 1: projection. 128 threads = 4 waves; wave w handles j-tile of 32 rows.
// grid.x = NB * 8  (8 j-tile groups of 4 per batch)
// ---------------------------------------------------------------------------
__global__ void __launch_bounds__(128)
gat_proj_kernel(const float* __restrict__ feat,   // [NB][NV][FIN]
                const float* __restrict__ W,      // [NH][FIN]
                const float* __restrict__ a,      // [2*NH]
                uint32_t*    __restrict__ hB,     // [NB][NJT][4][32][8] dwords (bf16 B-layout)
                float*       __restrict__ s_src,  // [NB][NV]
                float*       __restrict__ s_dst)  // [NB][NV]
{
    // row stride 66 floats: conflict-free for both the scatter-in and strided-out phases
    __shared__ float hT[4][32][66];

    const int tid  = threadIdx.x;
    const int w    = tid >> 5;
    const int lane = tid & 31;
    const int l16  = lane & 15;
    const int hi   = lane >> 4;

    const int n  = blockIdx.x >> 3;
    const int jt = ((blockIdx.x & 7) << 2) + w;
    const int j0 = jt * JT;

    // --- GEMM: 2 M-tiles (rows j0..j0+31) x 4 N-tiles (cols 0..63), K=128 ---
    v8f C[2][4];
#pragma unroll
    for (int mt = 0; mt < 2; ++mt)
#pragma unroll
        for (int ct = 0; ct < 4; ++ct)
#pragma unroll
            for (int r = 0; r < 8; ++r) C[mt][ct][r] = 0.0f;

    // A (f32 16x4): lane l -> M=l%16, K = k0 + 2*hi + {0,1}  (two consecutive f's)
    const float* A0 = feat + ((size_t)(n * NV + j0 + l16) * FIN) + 2 * hi;
    const float* A1 = A0 + (size_t)16 * FIN;
    // B (f32 4x16): lane l -> N=l%16, K = k0 + 2*hi + {0,1};  B[k][c] = W[c][k]
#pragma unroll 4
    for (int k0 = 0; k0 < FIN; k0 += 4) {
        v2f av0 = *(const v2f*)(A0 + k0);
        v2f av1 = *(const v2f*)(A1 + k0);
#pragma unroll
        for (int ct = 0; ct < 4; ++ct) {
            v2f bv = *(const v2f*)(W + (size_t)(ct * 16 + l16) * FIN + k0 + 2 * hi);
            C[0][ct] = __builtin_amdgcn_wmma_f32_16x16x4_f32(
                false, av0, false, bv, (short)0, C[0][ct], false, false);
            C[1][ct] = __builtin_amdgcn_wmma_f32_16x16x4_f32(
                false, av1, false, bv, (short)0, C[1][ct], false, false);
        }
    }

    // --- C layout -> LDS h tile [32 rows j][64 cols c] ---
    // VGPR r of a C tile holds row (r + 8*hi), column l16 of that 16x16 tile.
#pragma unroll
    for (int mt = 0; mt < 2; ++mt)
#pragma unroll
        for (int ct = 0; ct < 4; ++ct)
#pragma unroll
            for (int r = 0; r < 8; ++r)
                hT[w][mt * 16 + r + 8 * hi][ct * 16 + l16] = C[mt][ct][r];
    __syncthreads();

    // --- s_src / s_dst: lane l handles row l of this j-tile ---
    {
        float ssrc = 0.0f, sdst = 0.0f;
#pragma unroll 8
        for (int c = 0; c < NH; ++c) {
            float hv = hT[w][lane][c];
            ssrc = __builtin_fmaf(hv, a[c], ssrc);
            sdst = __builtin_fmaf(hv, a[NH + c], sdst);
        }
        s_src[n * NV + j0 + lane] = ssrc;
        s_dst[n * NV + j0 + lane] = sdst;
    }

    // --- repack h to bf16 WMMA-B operand, this code's K order (A matches): lane = (hi,N), 16 consecutive K/lane ---
    // B (bf16 32x16): lane l, vgpr v, half h -> K = 16*hi + 2v + h, N = l16
#pragma unroll
    for (int ct = 0; ct < 4; ++ct) {
        uint32_t pk[8];
#pragma unroll
        for (int v = 0; v < 8; ++v) {
            float lo = hT[w][16 * hi + 2 * v    ][ct * 16 + l16];
            float hh = hT[w][16 * hi + 2 * v + 1][ct * 16 + l16];
            pk[v] = pack_bf16(lo, hh);
        }
        uint32_t* dst = hB + ((((size_t)n * NJT + jt) * 4 + ct) * 32 + lane) * 8;
        *(uint4*)(dst)     = make_uint4(pk[0], pk[1], pk[2], pk[3]);
        *(uint4*)(dst + 4) = make_uint4(pk[4], pk[5], pk[6], pk[7]);
    }
}

// ---------------------------------------------------------------------------
// Pass 2: fused flash-GAT attention. 128 threads = 4 waves; wave owns a
// 16-row i-tile. grid = (NB, NV/64).
// ---------------------------------------------------------------------------
__global__ void __launch_bounds__(128)
gat_attn_kernel(const uint32_t* __restrict__ hB,
                const float*    __restrict__ s_src,
                const float*    __restrict__ s_dst,
                float*          __restrict__ out)   // [NB][NV][NH]
{
    __shared__ float lsd[NV];
    __shared__ float red[128];

    const int tid  = threadIdx.x;
    const int w    = tid >> 5;
    const int lane = tid & 31;
    const int l16  = lane & 15;
    const int hi   = lane >> 4;

    const int n  = blockIdx.x;
    const int i0 = blockIdx.y * 64 + w * 16;

    // stage s_dst[n] into LDS and find its max (exact softmax max via monotone lrelu)
    float pmax = -__builtin_inff();
    for (int k = tid; k < NV; k += 128) {
        float v = s_dst[n * NV + k];
        lsd[k] = v;
        pmax = fmaxf(pmax, v);
    }
    red[tid] = pmax;
    __syncthreads();
#pragma unroll
    for (int s = 64; s > 0; s >>= 1) {
        if (tid < s) red[tid] = fmaxf(red[tid], red[tid + s]);
        __syncthreads();
    }
    const float maxt = red[0];

    const float si = s_src[n * NV + i0 + l16];   // this lane's row score
    const float mi = lrelu(si + maxt);           // exact row max of e

    BfTile ones;
#pragma unroll
    for (int v = 0; v < 8; ++v) ones.u[v] = 0x3F803F80u;  // bf16 1.0 pair

    v8f acc[4], den;
#pragma unroll
    for (int ct = 0; ct < 4; ++ct)
#pragma unroll
        for (int r = 0; r < 8; ++r) acc[ct][r] = 0.0f;
#pragma unroll
    for (int r = 0; r < 8; ++r) den[r] = 0.0f;

    const uint32_t* hBn = hB + (size_t)n * NJT * 4 * 32 * 8;

    for (int jt = 0; jt < NJT; ++jt) {
        const int j0 = jt * JT;

        // A (bf16 16x32): lane l, vgpr v, half h -> M=l16, K=16*hi+2v+h
        BfTile A;
#pragma unroll
        for (int v = 0; v < 8; ++v) {
            const int K0 = (v << 1) + (hi << 4);
            float e0 = lrelu(si + lsd[j0 + K0]);
            float e1 = lrelu(si + lsd[j0 + K0 + 1]);
            A.u[v] = pack_bf16(__expf(e0 - mi), __expf(e1 - mi));
        }

#pragma unroll
        for (int ct = 0; ct < 4; ++ct) {
            BfTile B;
            const uint32_t* src = hBn + (((size_t)jt * 4 + ct) * 32 + lane) * 8;
            *(uint4*)&B.u[0] = *(const uint4*)(src);
            *(uint4*)&B.u[4] = *(const uint4*)(src + 4);
            acc[ct] = __builtin_amdgcn_wmma_f32_16x16x32_bf16(
                false, A.v, false, B.v, (short)0, acc[ct], false, false);
        }
        // denominator: row-sums of A land row-aligned with acc's C layout
        den = __builtin_amdgcn_wmma_f32_16x16x32_bf16(
            false, A.v, false, ones.v, (short)0, den, false, false);
    }

    // epilogue: C vgpr r -> row i0 + r + 8*hi, col ct*16 + l16
#pragma unroll
    for (int r = 0; r < 8; ++r) {
        const float inv = 1.0f / den[r];
        const int row = i0 + r + 8 * hi;
#pragma unroll
        for (int ct = 0; ct < 4; ++ct)
            out[((size_t)n * NV + row) * NH + ct * 16 + l16] = acc[ct][r] * inv;
    }
}

// ---------------------------------------------------------------------------
extern "C" void kernel_launch(void* const* d_in, const int* in_sizes, int n_in,
                              void* d_out, int out_size, void* d_ws, size_t ws_size,
                              hipStream_t stream) {
    (void)in_sizes; (void)n_in; (void)out_size; (void)ws_size;
    const float* feat = (const float*)d_in[0];   // 32*1024*128
    const float* W    = (const float*)d_in[1];   // 64*128
    const float* a    = (const float*)d_in[2];   // 128
    float* out        = (float*)d_out;           // 32*1024*64

    // workspace layout
    char* ws = (char*)d_ws;
    uint32_t* hB   = (uint32_t*)(ws);                           // 4 MB
    float*    ssrc = (float*)(ws + (size_t)4 * 1024 * 1024);    // 128 KB
    float*    sdst = (float*)(ws + (size_t)4 * 1024 * 1024 + 131072);

    gat_proj_kernel<<<dim3(NB * 8), dim3(128), 0, stream>>>(feat, W, a, hB, ssrc, sdst);
    gat_attn_kernel<<<dim3(NB, NV / 64), dim3(128), 0, stream>>>(hB, ssrc, sdst, out);
}
